// HetereoExpertFFN_57904749085343
// MI455X (gfx1250) — matrix operand order repaired and checked by reading, not compiled, not run on hardware
//
#include <hip/hip_runtime.h>
#include <hip/hip_bf16.h>
#include <math.h>

// ---------- types ----------
typedef __attribute__((ext_vector_type(16))) _Float16 v16h;
typedef __attribute__((ext_vector_type(8)))  float    v8f;
typedef __attribute__((ext_vector_type(4)))  float    f32x4;
typedef __attribute__((ext_vector_type(2)))  float    f32x2;

// ---------- problem constants ----------
#define DD    1024
#define HH    2944
#define EE    8
#define TT    4096
#define KT1   32      // DD/32   (K-tiles of GEMM1, f16 K=32)
#define NT1   184     // HH/16   (N-tiles of W1)
#define KT2   92      // HH/32   (K-tiles of GEMM2)
#define NT2   64      // DD/16   (N-tiles of W2)
#define XGROWS 8704   // padded dispatched rows: 2T + E*64
#define TPE   64      // max 64-token tiles per expert
#define HS    520     // LDS h-chunk row stride in f16 (512+8, spreads banks)
#define XS    1032    // LDS x-tile  row stride in f16 (1024+8)

// =====================================================================
// prep: tile/convert expert weights to f16 WMMA-B-fragment layout
// (W1h: [e][nt][kt][lane][16] so the k-loop streams contiguously;
//  W2h: [e][kt][nt][lane][16] so the n-loop streams contiguously),
// and gating W1 to f32 WMMA-B fragments.
// B-frag f16 (32x16): lane l -> col n = l&15, K = (l>>4)*16 + elem.
// B-frag f32 (4x16):  lane l -> col n = l&15, K = (l>>4)*2  + elem.
// =====================================================================
__global__ __launch_bounds__(256) void prep_weights(
    const float* __restrict__ eW1, const float* __restrict__ eW2,
    const float* __restrict__ gW1,
    _Float16* __restrict__ W1h, _Float16* __restrict__ W2h,
    float* __restrict__ gW1t)
{
  const int NW1 = EE*KT1*NT1*32;
  const int NW2 = EE*KT2*NT2*32;
  const int NG  = 256*64*32;
  int id = blockIdx.x*256 + threadIdx.x;
  if (id < NW1) {
    // dest layout: ((e*NT1 + nt)*KT1 + kt)*32 + lane
    int lane = id & 31; int r = id >> 5;
    int kt = r % KT1; r /= KT1;
    int nt = r % NT1; int e = r / NT1;
    int n  = nt*16 + (lane & 15);
    int k0 = kt*32 + (lane >> 4)*16;
    const float* src = eW1 + (size_t)e*DD*HH;
    v16h v;
#pragma unroll
    for (int i = 0; i < 16; ++i) v[i] = (_Float16)src[(size_t)(k0+i)*HH + n];
    *(v16h*)(W1h + (size_t)id*16) = v;
  } else if (id < NW1 + NW2) {
    // dest layout: ((e*KT2 + kt)*NT2 + nt)*32 + lane
    int id2 = id - NW1;
    int lane = id2 & 31; int r = id2 >> 5;
    int nt = r % NT2; r /= NT2;
    int kt = r % KT2; int e = r / KT2;
    int n  = nt*16 + (lane & 15);
    int k0 = kt*32 + (lane >> 4)*16;
    const float* src = eW2 + (size_t)e*HH*DD;
    v16h v;
#pragma unroll
    for (int i = 0; i < 16; ++i) v[i] = (_Float16)src[(size_t)(k0+i)*DD + n];
    *(v16h*)(W2h + (size_t)id2*16) = v;
  } else if (id < NW1 + NW2 + NG) {
    int id3 = id - NW1 - NW2;
    int lane = id3 & 31; int r = id3 >> 5;
    int nt = r & 63; int kt = r >> 6;
    int n  = nt*16 + (lane & 15);
    int k0 = kt*4 + (lane >> 4)*2;
    f32x2 v;
    v.x = gW1[(size_t)k0*DD + n];
    v.y = gW1[(size_t)(k0+1)*DD + n];
    *(f32x2*)(gW1t + (size_t)id3*2) = v;
  }
}

// x -> f32 WMMA-A-fragment layout for gating GEMM.
// A-frag f32 (16x4): lane l -> row = l&15, K = (l>>4)*2 + elem.
__global__ __launch_bounds__(256) void prep_xa(const float* __restrict__ x,
                                               float* __restrict__ xAf)
{
  int id = blockIdx.x*256 + threadIdx.x;          // (mt*256 + kt)*32 + lane
  if (id >= 256*256*32) return;
  int lane = id & 31; int r = id >> 5;
  int kt = r & 255; int mt = r >> 8;
  int row = mt*16 + (lane & 15);
  int k   = kt*4 + (lane >> 4)*2;
  f32x2 v = *(const f32x2*)(x + (size_t)row*DD + k);
  *(f32x2*)(xAf + (size_t)id*2) = v;
}

__global__ void zero_meta(int* meta) { if (threadIdx.x < 8) meta[threadIdx.x] = 0; }

// =====================================================================
// G1: hidden = relu(x @ gW1[:1024] + gb1 + snr*gW1[1024])  -- full fp32
// via v_wmma_f32_16x16x4_f32. Block: 256 thr (8 waves), tile 64x128.
// =====================================================================
__global__ __launch_bounds__(256) void g1_kernel(
    const float* __restrict__ xAf, const float* __restrict__ gW1t,
    const float* __restrict__ gW1, const float* __restrict__ gb1,
    const float* __restrict__ snr, float* __restrict__ hidden)
{
  int tid = threadIdx.x; int lane = tid & 31; int w = tid >> 5;
  int mq = w >> 1, nq = w & 1;
  int l15 = lane & 15, lh = lane >> 4;
  int mt_g = blockIdx.x*4 + mq;
  const f32x2* ap = (const f32x2*)xAf + (size_t)mt_g*256*32 + lane;
  v8f acc[4];
#pragma unroll
  for (int j = 0; j < 4; ++j) acc[j] = v8f{};
  for (int kt = 0; kt < 256; ++kt) {
    f32x2 a = ap[(size_t)kt*32];
#pragma unroll
    for (int j = 0; j < 4; ++j) {
      int ntg = blockIdx.y*8 + nq*4 + j;
      f32x2 b = ((const f32x2*)gW1t)[((size_t)kt*64 + ntg)*32 + lane];
      acc[j] = __builtin_amdgcn_wmma_f32_16x16x4_f32(
          false, a, false, b, (short)0, acc[j], false, false);
    }
  }
  float s = snr[0];
#pragma unroll
  for (int j = 0; j < 4; ++j) {
    int ntg = blockIdx.y*8 + nq*4 + j;
    int n = ntg*16 + l15;
    float bias = gb1[n] + s * gW1[(size_t)DD*DD + n];
#pragma unroll
    for (int v = 0; v < 8; ++v) {
      int row = blockIdx.x*64 + mq*16 + v + 8*lh;
      float hv = acc[j][v] + bias;
      hidden[(size_t)row*DD + n] = hv > 0.f ? hv : 0.f;
    }
  }
}

// =====================================================================
// G2: logits = hidden@gW2 + gb2; gumbel softmax; top-2; write
// gate_scores / expert_mask (outputs) and routing weights (ws).
// One wave per token.
// =====================================================================
__global__ __launch_bounds__(256) void g2_kernel(
    const float* __restrict__ hidden, const float* __restrict__ gW2,
    const float* __restrict__ gb2, const float* __restrict__ gu,
    float* __restrict__ gates_out, float* __restrict__ mask_out,
    float* __restrict__ top_w, int* __restrict__ top_e)
{
  int lane = threadIdx.x & 31; int w = threadIdx.x >> 5;
  int t = blockIdx.x*8 + w;
  float ps[8];
#pragma unroll
  for (int e = 0; e < 8; ++e) ps[e] = 0.f;
  const float* hrow = hidden + (size_t)t*DD;
  for (int i = 0; i < 32; ++i) {
    int k = i*32 + lane;
    float hv = hrow[k];
    const float* wr = gW2 + (size_t)k*EE;
#pragma unroll
    for (int e = 0; e < 8; ++e) ps[e] += hv * wr[e];
  }
#pragma unroll
  for (int e = 0; e < 8; ++e)
    for (int off = 16; off; off >>= 1) ps[e] += __shfl_xor(ps[e], off, 32);
  if (lane == 0) {
    float z[8], m = -1e30f, sc[8];
#pragma unroll
    for (int e = 0; e < 8; ++e) {
      float g = -logf(-logf(gu[t*8 + e] + 1e-9f) + 1e-9f);
      z[e] = ps[e] + gb2[e] + g;
      m = fmaxf(m, z[e]);
    }
    float sum = 0.f;
#pragma unroll
    for (int e = 0; e < 8; ++e) { sc[e] = expf(z[e] - m); sum += sc[e]; }
    float inv = 1.f / sum;
#pragma unroll
    for (int e = 0; e < 8; ++e) { sc[e] *= inv; gates_out[t*8 + e] = sc[e]; }
    int i1 = 0; float s1 = sc[0];
#pragma unroll
    for (int e = 1; e < 8; ++e) if (sc[e] > s1) { s1 = sc[e]; i1 = e; }
    int i2 = -1; float s2 = -1.f;
#pragma unroll
    for (int e = 0; e < 8; ++e) if (e != i1 && sc[e] > s2) { s2 = sc[e]; i2 = e; }
#pragma unroll
    for (int e = 0; e < 8; ++e) mask_out[t*8 + e] = (e == i1 || e == i2) ? 1.f : 0.f;
    top_w[t*2] = s1; top_w[t*2 + 1] = s2;
    top_e[t*2] = i1; top_e[t*2 + 1] = i2;
  }
}

// routing: per (token, k) slot assignment via atomics (order-free result)
__global__ void route_kernel(const int* __restrict__ top_e, int* meta,
                             int* __restrict__ listE, int* __restrict__ slot_p)
{
  int id = blockIdx.x*256 + threadIdx.x;
  if (id >= TT*2) return;
  int e = top_e[id];
  int pos = atomicAdd(&meta[e], 1);
  slot_p[id] = pos;
  listE[e*TT + pos] = id >> 1;
}

__global__ void offsets_kernel(int* meta)
{
  if (threadIdx.x == 0 && blockIdx.x == 0) {
    int acc = 0;
    for (int e = 0; e < 8; ++e) {
      int tl = (meta[e] + 63) >> 6;
      meta[8 + e]  = tl;
      meta[16 + e] = acc;
      acc += tl*64;
    }
  }
}

// =====================================================================
// Fused expert FFN, one block per (expert, 64-token tile).
// 512 thr = 16 waves. Token rows gathered straight from x into LDS f16.
// Chunked hidden (5x512 + 384):
//   phase1: h = relu(x @ W1 + b1) -> LDS. Waves 4Mx4N, k-outer with
//           A fragment held; per-wave C tiles cj[<=8]; W1 frags unique.
//   phase2: acc += h @ W2. Each wave owns a unique 64-col strip across
//           all 64 rows (16 C tiles) -> every W2 fragment load unique.
// All matrix math via v_wmma_f32_16x16x32_f16, f32 accumulation.
// =====================================================================
__global__ __launch_bounds__(512) void ffn_kernel(
    const float* __restrict__ x, const int* __restrict__ listE,
    const _Float16* __restrict__ W1h, const _Float16* __restrict__ W2h,
    const float* __restrict__ eb1, const float* __restrict__ eb2,
    const int* __restrict__ meta, float* __restrict__ yslot)
{
  __shared__ __align__(16) _Float16 sx[64*XS];   // 132 KB
  __shared__ __align__(16) _Float16 sh[64*HS];   // 66.5 KB
  int e = blockIdx.x / TPE; int tile = blockIdx.x % TPE;
  if (tile >= meta[8 + e]) return;               // block-uniform exit
  int cnt = meta[e]; int rowoff = meta[16 + e];
  int tid = threadIdx.x; int lane = tid & 31; int w = tid >> 5;
  int mq = w >> 2, nq = w & 3, ng = w;
  int l15 = lane & 15, lh = lane >> 4;

  // ---- stage 64 gathered token rows into LDS as f16 (row-major) ----
  {
    int row = tid >> 3;
    int seg = (tid & 7) * 128;
    int s = tile*64 + row;
    _Float16* dst = sx + row*XS + seg;
    if (s < cnt) {
      const float* src = x + (size_t)listE[e*TT + s]*DD + seg;
#pragma unroll
      for (int i = 0; i < 32; ++i) {
        f32x4 v = *(const f32x4*)(src + i*4);
        union { _Float16 h[4]; f32x2 d; } u;
        u.h[0] = (_Float16)v.x; u.h[1] = (_Float16)v.y;
        u.h[2] = (_Float16)v.z; u.h[3] = (_Float16)v.w;
        *(f32x2*)(dst + i*4) = u.d;
      }
    } else {
      f32x2 z = f32x2{};
#pragma unroll
      for (int i = 0; i < 32; ++i) *(f32x2*)(dst + i*4) = z;
    }
  }
  __syncthreads();

  const v16h* w1base = (const v16h*)W1h + (size_t)e*NT1*KT1*32 + lane;
  const v16h* w2base = (const v16h*)W2h + (size_t)e*KT2*NT2*32 + lane;

  v8f acc[16];
#pragma unroll
  for (int j = 0; j < 16; ++j) acc[j] = v8f{};

  for (int hc0 = 0; hc0 < HH; hc0 += 512) {
    int Hc = (HH - hc0) < 512 ? (HH - hc0) : 512;
    int NT4 = Hc >> 6;    // phase-1 tiles per wave: 8 or 6
    int KTC = Hc >> 5;    // phase-2 K tiles: 16 or 12

    // ---- phase 1: h_chunk = relu(x @ W1[:,chunk] + b1) -> LDS ----
    v8f cj[8];
#pragma unroll
    for (int jj = 0; jj < 8; ++jj) cj[jj] = v8f{};
    {
      const _Float16* sxr = sx + (mq*16 + l15)*XS + lh*16;
      for (int kt = 0; kt < KT1; ++kt) {
        union { v16h h; f32x4 f[2]; } a;
        a.f[0] = *(const f32x4*)(sxr + kt*32);
        a.f[1] = *(const f32x4*)(sxr + kt*32 + 8);
        const v16h* bp = w1base + ((size_t)((hc0 >> 4) + nq)*KT1 + kt)*32;
#pragma unroll
        for (int jj = 0; jj < 8; ++jj) {
          if (jj < NT4) {
            v16h b = bp[(size_t)jj*4*KT1*32];
            cj[jj] = __builtin_amdgcn_wmma_f32_16x16x32_f16(
                false, a.h, false, b, (short)0, cj[jj], false, false);
          }
        }
      }
    }
#pragma unroll
    for (int jj = 0; jj < 8; ++jj) {
      if (jj < NT4) {
        int ntg = (hc0 >> 4) + nq + jj*4;
        float b1v = eb1[e*HH + ntg*16 + l15];
        int nl = (nq + jj*4)*16 + l15;
        _Float16* shw = sh + (mq*16 + 8*lh)*HS + nl;
#pragma unroll
        for (int v = 0; v < 8; ++v) {
          float hv = cj[jj][v] + b1v;
          shw[v*HS] = (_Float16)(hv > 0.f ? hv : 0.f);
        }
      }
    }
    __syncthreads();

    // ---- phase 2: acc += h_chunk @ W2[chunk, :] ----
    for (int kt = 0; kt < KTC; ++kt) {
      union { v16h h; f32x4 f[2]; } am[4];
#pragma unroll
      for (int mt = 0; mt < 4; ++mt) {
        const _Float16* shr = sh + (mt*16 + l15)*HS + kt*32 + lh*16;
        am[mt].f[0] = *(const f32x4*)(shr);
        am[mt].f[1] = *(const f32x4*)(shr + 8);
      }
      const v16h* bp2 = w2base + (((size_t)(hc0 >> 5) + kt)*NT2 + ng*4)*32;
#pragma unroll
      for (int j = 0; j < 4; ++j) {
        v16h b = bp2[(size_t)j*32];
#pragma unroll
        for (int mt = 0; mt < 4; ++mt)
          acc[mt*4 + j] = __builtin_amdgcn_wmma_f32_16x16x32_f16(
              false, am[mt].h, false, b, (short)0, acc[mt*4 + j], false, false);
      }
    }
    __syncthreads();
  }

  // ---- epilogue: per-slot y = acc + b2 (weighting done in combine) ----
#pragma unroll
  for (int j = 0; j < 4; ++j) {
    int n = ng*64 + j*16 + l15;
    float b2v = eb2[e*DD + n];
#pragma unroll
    for (int mt = 0; mt < 4; ++mt) {
      int slotbase = rowoff + tile*64 + mt*16 + 8*lh;
#pragma unroll
      for (int v = 0; v < 8; ++v)
        yslot[(size_t)(slotbase + v)*DD + n] = acc[mt*4 + j][v] + b2v;
    }
  }
}

// combine: out[t] = w1*y[slot1] + w2*y[slot2]  (fully overwrites out)
__global__ __launch_bounds__(256) void combine_kernel(
    const float* __restrict__ yslot, const float* __restrict__ top_w,
    const int* __restrict__ top_e, const int* __restrict__ slot_p,
    const int* __restrict__ meta, float* __restrict__ out)
{
  int t = blockIdx.x;
  int e0 = top_e[t*2], e1 = top_e[t*2 + 1];
  float w0 = top_w[t*2], w1 = top_w[t*2 + 1];
  size_t g0 = (size_t)(meta[16 + e0] + slot_p[t*2]);
  size_t g1 = (size_t)(meta[16 + e1] + slot_p[t*2 + 1]);
  int c = threadIdx.x*4;
  f32x4 y0 = *(const f32x4*)(yslot + g0*DD + c);
  f32x4 y1 = *(const f32x4*)(yslot + g1*DD + c);
  f32x4 r = y0*w0 + y1*w1;
  *(f32x4*)(out + (size_t)t*DD + c) = r;
}

// =====================================================================
extern "C" void kernel_launch(void* const* d_in, const int* in_sizes, int n_in,
                              void* d_out, int out_size, void* d_ws, size_t ws_size,
                              hipStream_t stream) {
  (void)in_sizes; (void)n_in; (void)out_size; (void)ws_size;
  const float* x   = (const float*)d_in[0];
  const float* snr = (const float*)d_in[1];
  const float* gu  = (const float*)d_in[2];
  const float* gW1 = (const float*)d_in[3];
  const float* gb1 = (const float*)d_in[4];
  const float* gW2 = (const float*)d_in[5];
  const float* gb2 = (const float*)d_in[6];
  const float* eW1 = (const float*)d_in[7];
  const float* eb1 = (const float*)d_in[8];
  const float* eW2 = (const float*)d_in[9];
  const float* eb2 = (const float*)d_in[10];

  float* out   = (float*)d_out;
  float* gates = out + (size_t)TT*DD;
  float* mask  = gates + (size_t)TT*EE;

  // workspace layout (256B aligned)
  char* base = (char*)d_ws;
  size_t off = 0;
  auto alloc = [&](size_t bytes) {
    size_t r = off; off = (off + bytes + 255) & ~(size_t)255; return r;
  };
  _Float16* W1h   = (_Float16*)(base + alloc((size_t)EE*DD*HH*2));
  _Float16* W2h   = (_Float16*)(base + alloc((size_t)EE*HH*DD*2));
  float*    yslot = (float*)   (base + alloc((size_t)XGROWS*DD*4));
  float*    hidden= (float*)   (base + alloc((size_t)TT*DD*4));
  float*    xAf   = (float*)   (base + alloc((size_t)TT*DD*4));
  float*    gW1t  = (float*)   (base + alloc((size_t)DD*DD*4));
  float*    top_w = (float*)   (base + alloc((size_t)TT*2*4));
  int*      top_e = (int*)     (base + alloc((size_t)TT*2*4));
  int*      slot_p= (int*)     (base + alloc((size_t)TT*2*4));
  int*      listE = (int*)     (base + alloc((size_t)EE*TT*4));
  int*      meta  = (int*)     (base + alloc((size_t)24*4));

  const int NPREP = EE*KT1*NT1*32 + EE*KT2*NT2*32 + 256*64*32;
  prep_weights<<<(NPREP + 255)/256, 256, 0, stream>>>(eW1, eW2, gW1, W1h, W2h, gW1t);
  prep_xa<<<(256*256*32)/256, 256, 0, stream>>>(x, xAf);
  zero_meta<<<1, 32, 0, stream>>>(meta);
  g1_kernel<<<dim3(TT/64, DD/128), 256, 0, stream>>>(xAf, gW1t, gW1, gb1, snr, hidden);
  g2_kernel<<<TT/8, 256, 0, stream>>>(hidden, gW2, gb2, gu, gates, mask, top_w, top_e);
  route_kernel<<<(TT*2)/256, 256, 0, stream>>>(top_e, meta, listE, slot_p);
  offsets_kernel<<<1, 1, 0, stream>>>(meta);
  ffn_kernel<<<EE*TPE, 512, 0, stream>>>(x, listE, W1h, W2h, eb1, eb2, meta, yslot);
  combine_kernel<<<TT, 256, 0, stream>>>(yslot, top_w, top_e, slot_p, meta, out);
}
